// MultiHeadAtten_39702677684962
// MI455X (gfx1250) — matrix operand order repaired and checked by reading, not compiled, not run on hardware
//
#include <hip/hip_runtime.h>

// ---------------------------------------------------------------------------
// MI455X (gfx1250) multi-head attention forward.
//  - all matmuls on v_wmma_f32_16x16x32_bf16 (wave32 WMMA, f32 accum)
//  - fused fp32->bf16 conversion in GEMM LDS staging, software-pipelined loads
//  - final GEMM A-tile staged by the Tensor Data Mover (tensor_load_to_lds,
//    TDM pad feature reproduces the LDS bank padding), s_wait_tensorcnt sync
//  - flash attention: 64-col kv blocks, 16 WMMA per block, online softmax,
//    V pre-transposed by the V-projection epilogue
// Workspace: 64 MB (Qp, Kp, Vt, Xattn in bf16).
// ---------------------------------------------------------------------------

typedef __bf16 bf16_t;
typedef __attribute__((ext_vector_type(16))) bf16_t v16bf;
typedef __attribute__((ext_vector_type(8)))  bf16_t v8bf;
typedef __attribute__((ext_vector_type(8)))  float  v8f;
typedef __attribute__((ext_vector_type(4)))  unsigned int su4;
typedef __attribute__((ext_vector_type(8)))  unsigned int su8;

#define BATCH 4
#define SEQ   2048
#define DMODEL 1024
#define NHEAD 16
#define DK    64

#define BM 128
#define BN 128
#define BK 32

__device__ __forceinline__ v16bf cat16(v8bf lo, v8bf hi) {
  return __builtin_shufflevector(lo, hi, 0,1,2,3,4,5,6,7,8,9,10,11,12,13,14,15);
}

__device__ __forceinline__ v8f wmma_bf16(v16bf a, v16bf b, v8f c) {
  return __builtin_amdgcn_wmma_f32_16x16x32_bf16(
      false, a, false, b, (short)0, c, false, false);
}

__device__ __forceinline__ v8bf cvt8(float4 a, float4 b) {
  v8bf r;
  r[0]=(bf16_t)a.x; r[1]=(bf16_t)a.y; r[2]=(bf16_t)a.z; r[3]=(bf16_t)a.w;
  r[4]=(bf16_t)b.x; r[5]=(bf16_t)b.y; r[6]=(bf16_t)b.z; r[7]=(bf16_t)b.w;
  return r;
}

// ---------------------------------------------------------------------------
// GEMM: Out[m,n] = sum_k A[m,k]*W[n,k] + bias[n]
// AFMT 0: A fp32 (convert while staging, pipelined register prefetch)
// AFMT 1: A bf16 (staged by TDM tensor_load_to_lds with hardware padding)
// EPI 0: fp32 [M,N] | 1: bf16 head-major [bh][s][d] | 2: bf16 [bh][d][s]
// ---------------------------------------------------------------------------
template <int AFMT, int EPI>
__global__ __launch_bounds__(256)
void gemm_bf16_wmma(const void* __restrict__ Aptr, const float* __restrict__ Wf,
                    const float* __restrict__ bias, void* __restrict__ Out,
                    int M, int K, int N) {
  __shared__ bf16_t As[BM][BK + 8];   // 128 x 40 halfs  (TDM pad-compatible)
  __shared__ bf16_t Bs[BN][BK + 8];

  const int tid  = threadIdx.x;
  const int lane = tid & 31;
  const int wave = tid >> 5;
  const int ln   = lane & 15;
  const int hl   = lane >> 4;
  const int wm   = wave & 1;
  const int wn   = wave >> 1;

  const int n0 = blockIdx.x * BN + wn * 32;
  const int m0 = blockIdx.y * BM + wm * 64;

  const int lrow = tid >> 1;
  const int lcol = (tid & 1) << 4;
  const size_t aRow = (size_t)blockIdx.y * BM + lrow;
  const size_t bRow = (size_t)blockIdx.x * BN + lrow;

  v8f acc[4][2];
  for (int nt = 0; nt < 2; ++nt) {
    float bv = bias[n0 + nt * 16 + ln];
    for (int mt = 0; mt < 4; ++mt)
      for (int r = 0; r < 8; ++r) acc[mt][nt][r] = bv;
  }

  // ---- TDM descriptor for the bf16 A tile (block-uniform, ISA 8.3/8.4) ----
  su4 g0; su8 g1;
  unsigned long long gbase = 0;
  if constexpr (AFMT == 1) {
    gbase = (unsigned long long)(const bf16_t*)Aptr +
            ((size_t)blockIdx.y * BM) * K * sizeof(bf16_t);
    unsigned lds_addr = (unsigned)(size_t)&As[0][0];
    g0[0] = 1u;                       // count=1 descriptor
    g0[1] = lds_addr;                 // lds_addr [63:32]
    g0[2] = 0; g0[3] = 0x80000000u;   // type=2 (image) in [127:126]
    // data_size=2B, pad_enable, pad every 16 DWORDs (enc 3) by 4 DWORDs (enc 3)
    g1[0] = (1u << 16) | (1u << 20) | (3u << 22) | (3u << 25);
    g1[1] = (unsigned)(K & 0xFFFF) << 16;        // tensor_dim0 lo16 @ [63:48]
    g1[2] = (unsigned)(M & 0xFFFF) << 16;        // dim0 hi16=0, tensor_dim1 lo16
    g1[3] = ((unsigned)M >> 16) | (32u << 16);   // tensor_dim1 hi, tile_dim0=32
    g1[4] = 128u;                                // tile_dim1=128, tile_dim2=0
    g1[5] = (unsigned)K;                         // tensor_dim0_stride lo32
    g1[6] = 0; g1[7] = 0;
  }

  // ---- software-pipelined register staging ----
  float4 wa0, wa1, wa2, wa3;
  float4 aa0, aa1, aa2, aa3;
  uint4  ab0, ab1;
  const float* Wsrc = Wf + bRow * K + lcol;

  auto loadW = [&](int kk) {
    const float* s = Wsrc + kk;
    wa0 = ((const float4*)s)[0]; wa1 = ((const float4*)s)[1];
    wa2 = ((const float4*)s)[2]; wa3 = ((const float4*)s)[3];
  };
  auto loadA = [&](int kk) {
    if constexpr (AFMT == 0) {
      const float* s = (const float*)Aptr + aRow * K + kk + lcol;
      aa0 = ((const float4*)s)[0]; aa1 = ((const float4*)s)[1];
      aa2 = ((const float4*)s)[2]; aa3 = ((const float4*)s)[3];
    }
    (void)kk; (void)ab0; (void)ab1;
  };

  loadW(0); loadA(0);

  for (int k0 = 0; k0 < K; k0 += BK) {
    if constexpr (AFMT == 1) {
      if (wave == 0) {
        unsigned long long ga = gbase + (size_t)k0 * sizeof(bf16_t);
        g0[2] = (unsigned)ga;
        g0[3] = ((unsigned)(ga >> 32) & 0x1FFFFFFu) | 0x80000000u;
        asm volatile("tensor_load_to_lds %0, %1" :: "s"(g0), "s"(g1) : "memory");
      }
    } else {
      *(v8bf*)&As[lrow][lcol]     = cvt8(aa0, aa1);
      *(v8bf*)&As[lrow][lcol + 8] = cvt8(aa2, aa3);
    }
    *(v8bf*)&Bs[lrow][lcol]     = cvt8(wa0, wa1);
    *(v8bf*)&Bs[lrow][lcol + 8] = cvt8(wa2, wa3);

    if (k0 + BK < K) { loadW(k0 + BK); loadA(k0 + BK); }  // overlap with WMMA

    if constexpr (AFMT == 1) {
      if (wave == 0) __builtin_amdgcn_s_wait_tensorcnt(0);
    }
    __syncthreads();

    v16bf af[4], bfm[2];
#pragma unroll
    for (int mt = 0; mt < 4; ++mt) {
      int row = wm * 64 + mt * 16 + ln;
      int kh  = hl * 8;
      af[mt] = cat16(*(const v8bf*)&As[row][kh], *(const v8bf*)&As[row][kh + 16]);
    }
#pragma unroll
    for (int nt = 0; nt < 2; ++nt) {
      int col = wn * 32 + nt * 16 + ln;
      int kb  = hl * 8;
      bfm[nt] = cat16(*(const v8bf*)&Bs[col][kb], *(const v8bf*)&Bs[col][kb + 16]);
    }
#pragma unroll
    for (int mt = 0; mt < 4; ++mt)
#pragma unroll
      for (int nt = 0; nt < 2; ++nt)
        acc[mt][nt] = wmma_bf16(af[mt], bfm[nt], acc[mt][nt]);
    __syncthreads();
  }

#pragma unroll
  for (int mt = 0; mt < 4; ++mt)
#pragma unroll
    for (int nt = 0; nt < 2; ++nt)
#pragma unroll
      for (int r = 0; r < 8; ++r) {
        int gm = m0 + mt * 16 + hl * 8 + r;
        int gn = n0 + nt * 16 + ln;
        float v = acc[mt][nt][r];
        if constexpr (EPI == 0) {
          ((float*)Out)[(size_t)gm * N + gn] = v;
        } else {
          int b = gm >> 11, s = gm & (SEQ - 1);
          int h = gn >> 6,  d = gn & (DK - 1);
          bf16_t* Ob = (bf16_t*)Out;
          if constexpr (EPI == 1)
            Ob[((size_t)(b * NHEAD + h) * SEQ + s) * DK + d] = (bf16_t)v;
          else
            Ob[((size_t)(b * NHEAD + h) * DK + d) * SEQ + s] = (bf16_t)v;
        }
      }
}

// ---------------------------------------------------------------------------
// Flash attention: one wave = one (b,h, 16-row q tile); kv blocks of 64.
// 16 WMMA per block (8 QK^T + 8 PV), butterfly row-reductions amortized 2x.
// ---------------------------------------------------------------------------
__global__ __launch_bounds__(256)
void attn_flash(const bf16_t* __restrict__ Qp, const bf16_t* __restrict__ Kp,
                const bf16_t* __restrict__ Vt, bf16_t* __restrict__ Xa) {
  __shared__ bf16_t Ps[8][16][72];  // per-wave 16x64 (padded) P / output tile

  const int lane = threadIdx.x & 31;
  const int wave = threadIdx.x >> 5;
  const int ln = lane & 15, hl = lane >> 4;

  const int tile = blockIdx.x * 8 + wave;
  const int bh = tile >> 7;
  const int qt = tile & 127;

  const bf16_t* Qb = Qp + (size_t)bh * SEQ * DK;
  const bf16_t* Kb = Kp + (size_t)bh * SEQ * DK;
  const bf16_t* Vb = Vt + (size_t)bh * DK * SEQ;

  v16bf qf[2];
  {
    const bf16_t* qr = Qb + (size_t)(qt * 16 + ln) * DK;
#pragma unroll
    for (int f = 0; f < 2; ++f) {
      int d0 = hl * 8 + f * 32;
      qf[f] = cat16(*(const v8bf*)(qr + d0), *(const v8bf*)(qr + d0 + 16));
    }
  }

  float mrow[8], lrow[8];
  v8f acc[4], vzero;
  for (int r = 0; r < 8; ++r) { mrow[r] = -3.0e38f; lrow[r] = 0.f; vzero[r] = 0.f; }
  for (int t = 0; t < 4; ++t)
    for (int r = 0; r < 8; ++r) acc[t][r] = 0.f;

  for (int j = 0; j < SEQ / 64; ++j) {
    const int kv0 = j * 64;

    // ---- S = Q K^T over 64 kv columns (4 tiles x K=64) ----
    v8f st[4];
#pragma unroll
    for (int nt = 0; nt < 4; ++nt) {
      const bf16_t* kr = Kb + (size_t)(kv0 + nt * 16 + ln) * DK;
      int d0 = hl * 8;
      v16bf kf0 = cat16(*(const v8bf*)(kr + d0),      *(const v8bf*)(kr + d0 + 16));
      v16bf kf1 = cat16(*(const v8bf*)(kr + d0 + 32), *(const v8bf*)(kr + d0 + 48));
      st[nt] = wmma_bf16(qf[1], kf1, wmma_bf16(qf[0], kf0, vzero));
    }

    // ---- online softmax ----
    float cur[8];
#pragma unroll
    for (int r = 0; r < 8; ++r) {
      st[0][r] *= 0.125f; st[1][r] *= 0.125f;
      st[2][r] *= 0.125f; st[3][r] *= 0.125f;
      cur[r] = fmaxf(fmaxf(st[0][r], st[1][r]), fmaxf(st[2][r], st[3][r]));
    }
    for (int msk = 1; msk < 16; msk <<= 1)
#pragma unroll
      for (int r = 0; r < 8; ++r)
        cur[r] = fmaxf(cur[r], __shfl_xor(cur[r], msk, 32));
    float sc[8];
#pragma unroll
    for (int r = 0; r < 8; ++r) {
      float mn = fmaxf(mrow[r], cur[r]);
      sc[r] = __expf(mrow[r] - mn);
      mrow[r] = mn;
    }
    float p[4][8], rs[8];
#pragma unroll
    for (int r = 0; r < 8; ++r) {
      p[0][r] = __expf(st[0][r] - mrow[r]);
      p[1][r] = __expf(st[1][r] - mrow[r]);
      p[2][r] = __expf(st[2][r] - mrow[r]);
      p[3][r] = __expf(st[3][r] - mrow[r]);
      rs[r] = (p[0][r] + p[1][r]) + (p[2][r] + p[3][r]);
    }
    for (int msk = 1; msk < 16; msk <<= 1)
#pragma unroll
      for (int r = 0; r < 8; ++r)
        rs[r] += __shfl_xor(rs[r], msk, 32);
#pragma unroll
    for (int r = 0; r < 8; ++r) lrow[r] = lrow[r] * sc[r] + rs[r];
#pragma unroll
    for (int t = 0; t < 4; ++t)
#pragma unroll
      for (int r = 0; r < 8; ++r) acc[t][r] *= sc[r];

    // ---- P: C-layout -> LDS -> two A-fragments (K=0..31, 32..63) ----
#pragma unroll
    for (int nt = 0; nt < 4; ++nt)
#pragma unroll
      for (int r = 0; r < 8; ++r)
        Ps[wave][hl * 8 + r][nt * 16 + ln] = (bf16_t)p[nt][r];
    asm volatile("s_wait_dscnt 0x0" ::: "memory");
    v16bf pf0 = cat16(*(const v8bf*)&Ps[wave][ln][hl * 8],
                      *(const v8bf*)&Ps[wave][ln][hl * 8 + 16]);
    v16bf pf1 = cat16(*(const v8bf*)&Ps[wave][ln][32 + hl * 8],
                      *(const v8bf*)&Ps[wave][ln][32 + hl * 8 + 16]);

    // ---- acc += P V (V pre-transposed: column d contiguous in s) ----
#pragma unroll
    for (int t = 0; t < 4; ++t) {
      const bf16_t* vr = Vb + (size_t)(t * 16 + ln) * SEQ + kv0 + hl * 8;
      v16bf vf0 = cat16(*(const v8bf*)vr,        *(const v8bf*)(vr + 16));
      v16bf vf1 = cat16(*(const v8bf*)(vr + 32), *(const v8bf*)(vr + 48));
      acc[t] = wmma_bf16(pf1, vf1, wmma_bf16(pf0, vf0, acc[t]));
    }
  }

  // ---- normalize, restage via LDS, coalesced b128 stores ----
  float inv[8];
#pragma unroll
  for (int r = 0; r < 8; ++r) inv[r] = 1.0f / lrow[r];
#pragma unroll
  for (int t = 0; t < 4; ++t)
#pragma unroll
    for (int r = 0; r < 8; ++r)
      Ps[wave][hl * 8 + r][t * 16 + ln] = (bf16_t)(acc[t][r] * inv[r]);
  asm volatile("s_wait_dscnt 0x0" ::: "memory");

  const int b = bh >> 4, h = bh & 15;
#pragma unroll
  for (int pss = 0; pss < 4; ++pss) {
    int c = lane + 32 * pss;       // 128 chunks: 16 rows x 8 segs of 16B
    int row = c >> 3, seg = c & 7;
    uint4 d = *(const uint4*)&Ps[wave][row][seg * 8];
    *(uint4*)(Xa + ((size_t)(b * SEQ + qt * 16 + row)) * DMODEL +
              h * DK + seg * 8) = d;
  }
}

// ---------------------------------------------------------------------------
extern "C" void kernel_launch(void* const* d_in, const int* in_sizes, int n_in,
                              void* d_out, int out_size, void* d_ws, size_t ws_size,
                              hipStream_t stream) {
  (void)in_sizes; (void)n_in; (void)out_size; (void)ws_size;
  const float* q  = (const float*)d_in[0];
  const float* k  = (const float*)d_in[1];
  const float* v  = (const float*)d_in[2];
  const float* Wq = (const float*)d_in[4];
  const float* bq = (const float*)d_in[5];
  const float* Wk = (const float*)d_in[6];
  const float* bk = (const float*)d_in[7];
  const float* Wv = (const float*)d_in[8];
  const float* bv = (const float*)d_in[9];
  const float* Wo = (const float*)d_in[10];
  const float* bo = (const float*)d_in[11];

  const int M = BATCH * SEQ;
  const size_t seg = (size_t)M * DMODEL * sizeof(bf16_t);  // 16 MB
  char* ws = (char*)d_ws;
  bf16_t* Qp = (bf16_t*)(ws);
  bf16_t* Kp = (bf16_t*)(ws + seg);
  bf16_t* Vt = (bf16_t*)(ws + 2 * seg);
  bf16_t* Xa = (bf16_t*)(ws + 3 * seg);

  dim3 gGrid(DMODEL / BN, M / BM);
  dim3 blk(256);

  gemm_bf16_wmma<0, 1><<<gGrid, blk, 0, stream>>>(q, Wq, bq, Qp, M, DMODEL, DMODEL);
  gemm_bf16_wmma<0, 1><<<gGrid, blk, 0, stream>>>(k, Wk, bk, Kp, M, DMODEL, DMODEL);
  gemm_bf16_wmma<0, 2><<<gGrid, blk, 0, stream>>>(v, Wv, bv, Vt, M, DMODEL, DMODEL);

  attn_flash<<<(BATCH * NHEAD * (SEQ / 16)) / 8, blk, 0, stream>>>(Qp, Kp, Vt, Xa);

  gemm_bf16_wmma<1, 0><<<gGrid, blk, 0, stream>>>(Xa, Wo, bo, d_out, M, DMODEL, DMODEL);
}
